// HybridNetwork_37082747634059
// MI455X (gfx1250) — compile-verified, matrix operand order checked by reading, not run on hardware
//
#include <hip/hip_runtime.h>
#include <hip/hip_bf16.h>

// ---------------------------------------------------------------------------
// HybridNetwork (two-tower GraphSAGE) for MI455X / gfx1250.
// Per-layer: fused [mean-agg | x_tgt] (n x 512) bf16 GEMM vs packed
// [Wl^T ; Wr^T] (512 x 256) via v_wmma_f32_16x16x32_bf16, f32 accum,
// fused bias (+ReLU on layers 0 and 2). GEMM block tile = 32(M) x 256(N),
// fully-unrolled software-pipelined K loop (kt+1 fragment loads issue under
// kt's WMMAs; full unroll makes the rotation pure SSA renaming -> no v_movs,
// no WMMA->VALU hazard NOPs).
// ---------------------------------------------------------------------------

#define HDIM 256
#define KDIM 512      // concat K (agg 256 + self 256)
#define FANOUT 10

typedef unsigned short u16;
typedef __attribute__((ext_vector_type(16))) __bf16 v16bf;
typedef __attribute__((ext_vector_type(8)))  float  v8f;

__device__ __forceinline__ u16 f32_to_bf16(float f) {
    unsigned u = __builtin_bit_cast(unsigned, f);
    unsigned r = u + 0x7fffu + ((u >> 16) & 1u);   // round-to-nearest-even
    return (u16)(r >> 16);
}

#define WMMA_BF16(A_, B_, C_) \
    __builtin_amdgcn_wmma_f32_16x16x32_bf16(false, (A_), false, (B_), (short)0, (C_), false, false)

// ---------------------------------------------------------------------------
// Repack Wcat[k, o] = (k < 256 ? Wl[o, k] : Wr[o, k-256]) into WMMA B-fragment
// order: Wp[((kt*16 + nt)*32 + lane)*16 + i] = Wcat[kt*32 + (lane>>4)*16 + i,
//                                                   nt*16 + (lane&15)]
// so each lane's 16 bf16 fragment elements are one contiguous 32B load.
// grid = 512 blocks x 256 threads (131072 fragment elements)
// ---------------------------------------------------------------------------
__global__ __launch_bounds__(256) void pack_w_kernel(
    const float* __restrict__ Wl, const float* __restrict__ Wr,
    u16* __restrict__ Wp)
{
    int idx  = blockIdx.x * 256 + threadIdx.x;
    int i    = idx & 15;
    int lane = (idx >> 4) & 31;
    int nt   = (idx >> 9) & 15;
    int kt   = (idx >> 13) & 15;
    int k = kt * 32 + (lane >> 4) * 16 + i;
    int o = nt * 16 + (lane & 15);
    float v = (k < HDIM) ? Wl[(size_t)o * HDIM + k]
                         : Wr[(size_t)o * HDIM + (k - HDIM)];
    Wp[idx] = f32_to_bf16(v);
}

// ---------------------------------------------------------------------------
// Build bf16 A rows: A[node, 0:256]  = 0.1 * sum of 10 gathered source rows
//                    A[node, 256:512]= x_tgt row (= first n_dst rows of x)
// grid = n_dst blocks x 256 threads (thread == feature)
// ---------------------------------------------------------------------------
__global__ __launch_bounds__(256) void agg_kernel(
    const float* __restrict__ x, const int* __restrict__ erow,
    u16* __restrict__ A)
{
    int node = blockIdx.x;
    int t    = threadIdx.x;
    const int* rows = erow + (size_t)node * FANOUT;   // uniform -> scalar loads

    // CDNA5 global_prefetch_b8 of the self row we read at the end
    __builtin_prefetch(&x[(size_t)node * HDIM + t], 0, 1);

    float acc = 0.f;
#pragma unroll
    for (int k = 0; k < FANOUT; ++k) {
        if (k + 1 < FANOUT)
            __builtin_prefetch(&x[(size_t)rows[k + 1] * HDIM + t], 0, 1);
        acc += x[(size_t)rows[k] * HDIM + t];
    }
    acc *= 0.1f;   // mean over exactly FANOUT=10 edges per target

    u16* arow = A + (size_t)node * KDIM;
    arow[t]        = f32_to_bf16(acc);
    arow[HDIM + t] = f32_to_bf16(x[(size_t)node * HDIM + t]);
}

// ---------------------------------------------------------------------------
// Pad the tail rows of A (up to the next multiple of 32) with zeros so the
// M-padded GEMM reads defined data. grid = pad_rows blocks x 256 threads.
// ---------------------------------------------------------------------------
__global__ __launch_bounds__(256) void zero_pad_kernel(
    u16* __restrict__ A, int first_row)
{
    size_t base = (size_t)(first_row + blockIdx.x) * KDIM + threadIdx.x;
    A[base] = 0;
    A[base + HDIM] = 0;
}

// ---------------------------------------------------------------------------
// C[n x 256] = A[n x 512]_bf16 @ Wp[512 x 256]_bf16 + bias (+ReLU), f32 out.
// Block = 128 threads = 4 waves; block tile = 32(M) x 256(N); each wave owns
// a 32 x 64 slice (8 accumulators). K: 16 steps of 32, fully unrolled with
// next-step fragments loaded before this step's WMMAs.
// A-fragment (16x32 bf16): lane<16 -> row=lane, K-low half; lane>=16 -> K-high.
// B-fragment (32x16 bf16): lane<16 -> col=lane, K-low half; lane>=16 -> K-high.
// D (16x16 f32): element v -> row = v + 8*(lane>=16), col = lane&15.
// ---------------------------------------------------------------------------
__global__ __launch_bounds__(128) void gemm_kernel(
    const u16* __restrict__ A, const u16* __restrict__ Wp,
    const float* __restrict__ bias, float* __restrict__ out,
    int n_dst, int relu)
{
    const int lane  = threadIdx.x & 31;
    const int wave  = threadIdx.x >> 5;     // 0..3 -> N slice
    const int mbase = blockIdx.x * 32;
    const int row   = lane & 15;
    const int khalf = lane >> 4;
    const int nt0   = wave * 4;

    v8f acc[8];
#pragma unroll
    for (int j = 0; j < 8; ++j) acc[j] = (v8f){};

    const u16* arow0 = A + (size_t)(mbase + row) * KDIM + khalf * 16;
    const u16* arow1 = arow0 + (size_t)16 * KDIM;
    const u16* wbase = Wp + (size_t)lane * 16;

    // ---- prologue: fragments for kt = 0 ----
    v16bf a0 = *reinterpret_cast<const v16bf*>(arow0);
    v16bf a1 = *reinterpret_cast<const v16bf*>(arow1);
    v16bf b[4];
    {
        const u16* wb = wbase + (size_t)nt0 * 512;
#pragma unroll
        for (int t = 0; t < 4; ++t)
            b[t] = *reinterpret_cast<const v16bf*>(wb + t * 512);
    }

    // ---- fully unrolled pipelined K loop ----
#pragma unroll
    for (int kt = 0; kt < 16; ++kt) {
        v16bf an0, an1, bn[4];
        if (kt < 15) {
            an0 = *reinterpret_cast<const v16bf*>(arow0 + (kt + 1) * 32);
            an1 = *reinterpret_cast<const v16bf*>(arow1 + (kt + 1) * 32);
            const u16* wb = wbase + (size_t)((kt + 1) * 16 + nt0) * 512;
#pragma unroll
            for (int t = 0; t < 4; ++t)
                bn[t] = *reinterpret_cast<const v16bf*>(wb + t * 512);
        }
#pragma unroll
        for (int t = 0; t < 4; ++t) {
            acc[t]     = WMMA_BF16(a0, b[t], acc[t]);
            acc[4 + t] = WMMA_BF16(a1, b[t], acc[4 + t]);
        }
        if (kt < 15) {
            a0 = an0; a1 = an1;
#pragma unroll
            for (int t = 0; t < 4; ++t) b[t] = bn[t];
        }
    }

    // ---- store (guard after all WMMAs; EXEC is all-ones during matrix ops) ----
    const int colL = lane & 15;
#pragma unroll
    for (int mt = 0; mt < 2; ++mt) {
        const int rowout = mbase + mt * 16 + 8 * khalf;
#pragma unroll
        for (int t = 0; t < 4; ++t) {
            int c = (nt0 + t) * 16 + colL;
            float bv = bias[c];
#pragma unroll
            for (int v = 0; v < 8; ++v) {
                int r = rowout + v;
                float val = acc[mt * 4 + t][v] + bv;
                if (relu) val = fmaxf(val, 0.f);
                if (r < n_dst)
                    out[(size_t)r * HDIM + c] = val;
            }
        }
    }
}

// ---------------------------------------------------------------------------
// Prefix offsets of num_users_per_res (B = 4096; O(B^2) scan is trivial).
// ---------------------------------------------------------------------------
__global__ __launch_bounds__(256) void offsets_kernel(
    const int* __restrict__ counts, int* __restrict__ off)
{
    int r = blockIdx.x * 256 + threadIdx.x;
    if (r >= 4096) return;
    int s = 0;
    for (int j = 0; j < r; ++j) s += counts[j];
    off[r] = s;
}

// ---------------------------------------------------------------------------
// pooled[r] = mean over users of user_out[inverse_idx[t]].
// grid = 4096 blocks x 256 threads (thread == feature)
// ---------------------------------------------------------------------------
__global__ __launch_bounds__(256) void pool_kernel(
    const float* __restrict__ uout, const int* __restrict__ inv,
    const int* __restrict__ counts, const int* __restrict__ off,
    float* __restrict__ pooled)
{
    int r = blockIdx.x;
    int t = threadIdx.x;
    int o = off[r];
    int c = counts[r];
    float acc = 0.f;
    for (int k = 0; k < c; ++k) {
        int u = inv[o + k];
        acc += uout[(size_t)u * HDIM + t];
    }
    pooled[(size_t)r * HDIM + t] = acc / (float)c;
}

// ---------------------------------------------------------------------------
// out[r, o] = b_out[o] + dot(concat(res_out[r], pooled[r]), W_out[o, :]).
// One wave per row; __shfl_xor wave32 reduction. grid = 512 x 256.
// ---------------------------------------------------------------------------
__global__ __launch_bounds__(256) void head_kernel(
    const float* __restrict__ rout, const float* __restrict__ pooled,
    const float* __restrict__ Wout, const float* __restrict__ bout,
    float* __restrict__ out)
{
    int wave = threadIdx.x >> 5;
    int lane = threadIdx.x & 31;
    int r = blockIdx.x * 8 + wave;
    float a0 = 0.f, a1 = 0.f;
    for (int k = lane; k < 2 * HDIM; k += 32) {
        float v = (k < HDIM) ? rout[(size_t)r * HDIM + k]
                             : pooled[(size_t)r * HDIM + (k - HDIM)];
        a0 += v * Wout[k];
        a1 += v * Wout[2 * HDIM + k];
    }
#pragma unroll
    for (int s = 16; s >= 1; s >>= 1) {
        a0 += __shfl_xor(a0, s, 32);
        a1 += __shfl_xor(a1, s, 32);
    }
    if (lane == 0) {
        out[(size_t)r * 2 + 0] = a0 + bout[0];
        out[(size_t)r * 2 + 1] = a1 + bout[1];
    }
}

// ---------------------------------------------------------------------------
extern "C" void kernel_launch(void* const* d_in, const int* in_sizes, int n_in,
                              void* d_out, int out_size, void* d_ws, size_t ws_size,
                              hipStream_t stream)
{
    (void)in_sizes; (void)n_in; (void)out_size; (void)ws_size;

    const float* res_x  = (const float*)d_in[0];
    const float* user_x = (const float*)d_in[1];
    const int* res_ei[3]  = {(const int*)d_in[2], (const int*)d_in[3], (const int*)d_in[4]};
    const int* user_ei[3] = {(const int*)d_in[5], (const int*)d_in[6], (const int*)d_in[7]};
    const int* counts = (const int*)d_in[8];
    const int* inv    = (const int*)d_in[9];
    const float* res_Wl[3] = {(const float*)d_in[10], (const float*)d_in[13], (const float*)d_in[16]};
    const float* res_bl[3] = {(const float*)d_in[11], (const float*)d_in[14], (const float*)d_in[17]};
    const float* res_Wr[3] = {(const float*)d_in[12], (const float*)d_in[15], (const float*)d_in[18]};
    const float* user_Wl[3] = {(const float*)d_in[19], (const float*)d_in[22], (const float*)d_in[25]};
    const float* user_bl[3] = {(const float*)d_in[20], (const float*)d_in[23], (const float*)d_in[26]};
    const float* user_Wr[3] = {(const float*)d_in[21], (const float*)d_in[24], (const float*)d_in[27]};
    const float* W_out = (const float*)d_in[28];
    const float* b_out = (const float*)d_in[29];

    const int RN[4] = {200000, 50000, 12000, 4096};
    const int UN[4] = {250000, 60000, 20000, 8192};

    // ---- carve workspace (256B-aligned sub-buffers) ----
    char* ws = (char*)d_ws;
    size_t pos = 0;
    auto carve = [&](size_t bytes) -> char* {
        char* p = ws + pos;
        pos += (bytes + 255) & ~(size_t)255;
        return p;
    };
    u16* wp[6];
    for (int l = 0; l < 6; ++l) wp[l] = (u16*)carve((size_t)KDIM * HDIM * 2);
    int*   offs    = (int*)carve(4096 * sizeof(int));
    u16*   Abuf    = (u16*)carve((size_t)60032 * KDIM * 2);   // padded to mult of 32
    float* bufA    = (float*)carve((size_t)60000 * HDIM * 4);
    float* bufB    = (float*)carve((size_t)60000 * HDIM * 4);
    float* resout  = (float*)carve((size_t)4096 * HDIM * 4);
    float* userout = (float*)carve((size_t)8192 * HDIM * 4);
    float* pooled  = (float*)carve((size_t)4096 * HDIM * 4);

    // ---- pack weights (6 layers) into WMMA B-fragment order ----
    for (int l = 0; l < 3; ++l)
        pack_w_kernel<<<512, 256, 0, stream>>>(res_Wl[l], res_Wr[l], wp[l]);
    for (int l = 0; l < 3; ++l)
        pack_w_kernel<<<512, 256, 0, stream>>>(user_Wl[l], user_Wr[l], wp[3 + l]);

    // ---- one GNN layer: gather/convert then WMMA GEMM (+bias/ReLU) ----
    auto layer = [&](const float* x_src, const int* erow, const float* bl,
                     u16* wpt, float* x_out, int n_dst, int relu) {
        agg_kernel<<<n_dst, 256, 0, stream>>>(x_src, erow, Abuf);
        int pad = ((n_dst + 31) & ~31) - n_dst;
        if (pad > 0)
            zero_pad_kernel<<<pad, 256, 0, stream>>>(Abuf, n_dst);
        gemm_kernel<<<(n_dst + 31) / 32, 128, 0, stream>>>(
            Abuf, wpt, bl, x_out, n_dst, relu);
    };

    auto tower = [&](const float* x0, const int* const* ei,
                     const float* const* bl, u16* const* wpt,
                     const int* N, float* outbuf) {
        layer(x0,   ei[0], bl[0], wpt[0], bufA,   N[1], 1);  // layer 0 (ReLU)
        layer(bufA, ei[1], bl[1], wpt[1], bufB,   N[2], 0);  // layer 1
        layer(bufB, ei[2], bl[2], wpt[2], outbuf, N[3], 1);  // layer 2 (ReLU)
    };
    u16* wpr[3] = {wp[0], wp[1], wp[2]};
    u16* wpu[3] = {wp[3], wp[4], wp[5]};
    tower(res_x,  res_ei,  res_bl,  wpr, RN, resout);
    tower(user_x, user_ei, user_bl, wpu, UN, userout);

    // ---- pooling + head ----
    offsets_kernel<<<16, 256, 0, stream>>>(counts, offs);
    pool_kernel<<<4096, 256, 0, stream>>>(userout, inv, counts, offs, pooled);
    head_kernel<<<512, 256, 0, stream>>>(resout, pooled, W_out, b_out, (float*)d_out);
}
